// TranslateModel_32315333935703
// MI455X (gfx1250) — matrix operand order repaired and checked by reading, not compiled, not run on hardware
//
#include <hip/hip_runtime.h>
#include <hip/hip_bf16.h>
#include <math.h>

// ---------------- model dims (match reference) ----------------
#define VOCAB_N 32000
#define DM      1024          // d_model
#define DI      2048          // d_inner
#define NL      4
#define DSTATE  8
#define DCONV   4
#define DTRANK  64
#define BB      2
#define LSEQ    1024
#define MROWS   (BB * LSEQ)   // 2048 token rows

// ---------------- WMMA GEMM tiling ----------------
#define BM 128
#define BN 128
#define BK 32
#define LDSS 40               // BK + 8 pad -> 80B row stride (16B aligned, bank-conflict-free)

typedef __attribute__((ext_vector_type(16))) __bf16 v16bf;
typedef __attribute__((ext_vector_type(8)))  __bf16 v8bf;
typedef __attribute__((ext_vector_type(4)))  __bf16 v4bf;
typedef __attribute__((ext_vector_type(8)))  float  v8f;

__device__ __forceinline__ float sigmoidf_(float x) { return 1.f / (1.f + __expf(-x)); }
__device__ __forceinline__ float softplusf_(float x) {
    return (x > 20.f) ? x : log1pf(__expf(x));
}
__device__ __forceinline__ int imin_(int a, int b) { return a < b ? a : b; }

// ======================================================================
// Tiled bf16 WMMA GEMM:  C[M,N] = A[M,K] (f32 row-major, stride lda)
//                               x W[N,K]^T (f32 row-major weights)
// mode: 0 = none, 1 = +bias[col], 2 = softplus(x + bias[col])
// f32 -> bf16 conversion happens while staging into LDS.
// 256 threads = 8 wave32; each wave computes a 32x64 sub-tile
// (2x4 tiles of v_wmma_f32_16x16x32_bf16).
//
// Software pipeline per k-step:
//   issue global loads (tile ks+1) -> 8x WMMA on LDS[cur]
//   -> wait/convert/ds_store into LDS[nxt] -> barrier
// so HBM latency is covered by matrix work.
// ======================================================================
__global__ __launch_bounds__(256)
void gemm_bf16wmma_kernel(const float* __restrict__ A, int lda,
                          const float* __restrict__ W,
                          float* __restrict__ C, int ldc,
                          const float* __restrict__ bias,
                          int M, int N, int K, int mode)
{
    __shared__ __align__(16) __bf16 lds_a[2][BM][LDSS];   // [m][k]
    __shared__ __align__(16) __bf16 lds_b[2][BN][LDSS];   // B^T: [n][k]

    const int tid  = threadIdx.x;
    const int lane = tid & 31;
    const int wave = tid >> 5;
    const int m0 = blockIdx.y * BM;
    const int n0 = blockIdx.x * BN;
    const int wm = (wave >> 1) * 32;   // wave row offset inside block tile
    const int wn = (wave & 1) * 64;    // wave col offset inside block tile

    // Per-thread staging coordinates: 4 float4 chunks per matrix per k-step.
    // chunk i -> row (tid>>3) + 32*i, k-offset (tid&7)*4  (constant across i)
    const int rr = tid >> 3;           // 0..31
    const int k4 = (tid & 7) * 4;

    // Clamped row indices (branch-free edge handling; cols >= N never stored)
    int ga_row[4], gb_row[4];
#pragma unroll
    for (int i = 0; i < 4; ++i) {
        ga_row[i] = imin_(m0 + rr + 32 * i, M - 1);
        gb_row[i] = imin_(n0 + rr + 32 * i, N - 1);
    }

    const v8f vzero = {0.f, 0.f, 0.f, 0.f, 0.f, 0.f, 0.f, 0.f};
    v8f acc[2][4];
#pragma unroll
    for (int i = 0; i < 2; ++i)
#pragma unroll
        for (int j = 0; j < 4; ++j) acc[i][j] = vzero;

    const int nk = K / BK;             // K is always a multiple of 32 here

    float4 ra[4], rb[4];               // in-flight staging registers

    auto issue_loads = [&](int ks) {
        const int k0 = ks * BK;
#pragma unroll
        for (int i = 0; i < 4; ++i)
            ra[i] = *(const float4*)(A + (size_t)ga_row[i] * lda + k0 + k4);
#pragma unroll
        for (int i = 0; i < 4; ++i)
            rb[i] = *(const float4*)(W + (size_t)gb_row[i] * K + k0 + k4);
        // prefetch tile ks+2 two stages ahead (global_prefetch_b8)
        const int k0p = k0 + BK;
        if (k0p + BK <= K) {
            __builtin_prefetch(A + (size_t)ga_row[0] * lda + k0p + k4, 0, 0);
            __builtin_prefetch(W + (size_t)gb_row[0] * K + k0p + k4, 0, 0);
        }
    };

    auto commit = [&](int buf) {
#pragma unroll
        for (int i = 0; i < 4; ++i) {
            v4bf p = { (__bf16)ra[i].x, (__bf16)ra[i].y,
                       (__bf16)ra[i].z, (__bf16)ra[i].w };
            *(v4bf*)&lds_a[buf][rr + 32 * i][k4] = p;
        }
#pragma unroll
        for (int i = 0; i < 4; ++i) {
            v4bf p = { (__bf16)rb[i].x, (__bf16)rb[i].y,
                       (__bf16)rb[i].z, (__bf16)rb[i].w };
            *(v4bf*)&lds_b[buf][rr + 32 * i][k4] = p;
        }
    };

    const int l15   = lane & 15;
    const int khalf = (lane >> 4) * 8;    // A: lanes16-31 start at K=8 (ISA layout)
    const int bg    = (lane >> 4) * 8;    // B: lanes16-31 start at K=8 (ISA layout)

    // prologue: stage tile 0
    issue_loads(0);
    commit(0);
    __syncthreads();

    for (int ks = 0; ks < nk; ++ks) {
        const int cur  = ks & 1;
        const bool more = (ks + 1 < nk);

        if (more) issue_loads(ks + 1);   // loads in flight during WMMAs

        v16bf afr[2], bfr[4];
#pragma unroll
        for (int mt = 0; mt < 2; ++mt) {
            const __bf16* ap = &lds_a[cur][wm + mt * 16 + l15][0];
            v8bf lo = *(const v8bf*)(ap + khalf);        // K khalf..khalf+7
            v8bf hi = *(const v8bf*)(ap + khalf + 16);   // K khalf+16..+23
            afr[mt] = __builtin_shufflevector(lo, hi, 0,1,2,3,4,5,6,7,
                                                      8,9,10,11,12,13,14,15);
        }
#pragma unroll
        for (int nt = 0; nt < 4; ++nt) {
            const __bf16* bp = &lds_b[cur][wn + nt * 16 + l15][0];
            v8bf lo = *(const v8bf*)(bp + bg);           // K bg..bg+7
            v8bf hi = *(const v8bf*)(bp + bg + 16);      // K bg+16..bg+23
            bfr[nt] = __builtin_shufflevector(lo, hi, 0,1,2,3,4,5,6,7,
                                                      8,9,10,11,12,13,14,15);
        }
#pragma unroll
        for (int mt = 0; mt < 2; ++mt)
#pragma unroll
            for (int nt = 0; nt < 4; ++nt)
                acc[mt][nt] = __builtin_amdgcn_wmma_f32_16x16x32_bf16(
                    false, afr[mt], false, bfr[nt],
                    (short)0, acc[mt][nt], false, false);

        if (more) commit(cur ^ 1);       // s_wait_loadcnt lands after the WMMAs
        __syncthreads();
    }

    // ---- epilogue: C layout per ISA (VGPR v -> row v + 8*(lane>>4), col lane&15)
#pragma unroll
    for (int mt = 0; mt < 2; ++mt)
#pragma unroll
        for (int nt = 0; nt < 4; ++nt) {
            int rbase = m0 + wm + mt * 16 + (lane >> 4) * 8;
            int col   = n0 + wn + nt * 16 + l15;
            if (col >= N) continue;
#pragma unroll
            for (int v = 0; v < 8; ++v) {
                int row = rbase + v;
                if (row >= M) continue;
                float val = acc[mt][nt][v];
                if (mode >= 1) val += bias[col];
                if (mode == 2) val = softplusf_(val);
                C[(size_t)row * ldc + col] = val;
            }
        }
}

// ======================================================================
// Embedding gather: x[m,:] = embed[ids[m],:]
// ======================================================================
__global__ void embed_kernel(const int* __restrict__ ids,
                             const float* __restrict__ emb,
                             float* __restrict__ x)
{
    int m = blockIdx.x;
    int id = ids[m];
    const float* src = emb + (size_t)id * DM;
    float* dst = x + (size_t)m * DM;
    for (int d = threadIdx.x; d < DM; d += blockDim.x) dst[d] = src[d];
}

// ======================================================================
// Causal depthwise conv1d (K=4) + bias + SiLU.  xz holds [xc | z] (2*DI cols).
// ======================================================================
__global__ void conv_silu_kernel(const float* __restrict__ xz,
                                 const float* __restrict__ cw,
                                 const float* __restrict__ cb,
                                 float* __restrict__ xc)
{
    int d = blockIdx.x * blockDim.x + threadIdx.x;   // channel
    int r = blockIdx.y;                              // b*L + l
    int l = r & (LSEQ - 1);
    float acc = cb[d];
#pragma unroll
    for (int t = 0; t < DCONV; ++t) {
        int back = (DCONV - 1) - t;
        if (l - back >= 0)
            acc += xz[(size_t)(r - back) * (2 * DI) + d] * cw[d * DCONV + t];
    }
    xc[(size_t)r * DI + d] = acc * sigmoidf_(acc);
}

// ======================================================================
// Selective scan: one lane per (b, d_inner) channel, h[8] in VGPRs,
// sequential over L.  Fused epilogue: y = (y + xc*D) * silu(z).
// ======================================================================
__global__ __launch_bounds__(256)
void scan_kernel(const float* __restrict__ dtf,   // (M, DI) softplus'ed delta
                 const float* __restrict__ xdbl,  // (M, 80): B @ 64..71, C @ 72..79
                 const float* __restrict__ xc,    // (M, DI)
                 const float* __restrict__ xz,    // (M, 2*DI), z at col DI+d
                 const float* __restrict__ Alog,  // (DI, 8)
                 const float* __restrict__ Dp,    // (DI)
                 float* __restrict__ y)           // (M, DI)
{
    int d = blockIdx.x * blockDim.x + threadIdx.x;
    int b = blockIdx.y;

    float Ad[DSTATE];
#pragma unroll
    for (int n = 0; n < DSTATE; ++n) Ad[n] = -__expf(Alog[(size_t)d * DSTATE + n]);
    const float Dd = Dp[d];

    float h[DSTATE];
#pragma unroll
    for (int n = 0; n < DSTATE; ++n) h[n] = 0.f;

    for (int l = 0; l < LSEQ; ++l) {
        const size_t r = (size_t)b * LSEQ + l;
        const float dtv = dtf[r * DI + d];
        const float xv  = xc[r * DI + d];
        const float* bc = xdbl + r * (DTRANK + 2 * DSTATE);
        float yv = 0.f;
#pragma unroll
        for (int n = 0; n < DSTATE; ++n) {
            float Bn = bc[DTRANK + n];
            float Cn = bc[DTRANK + DSTATE + n];
            float dA = __expf(dtv * Ad[n]);
            h[n] = fmaf(dA, h[n], dtv * Bn * xv);
            yv   = fmaf(h[n], Cn, yv);
        }
        float zv = xz[r * (2 * DI) + DI + d];
        y[r * DI + d] = (yv + xv * Dd) * (zv * sigmoidf_(zv));
    }
}

// ======================================================================
// (optional residual) + LayerNorm over DM=1024, one block per row.
// ======================================================================
__global__ __launch_bounds__(256)
void add_ln_kernel(const float* __restrict__ inp, const float* __restrict__ res,
                   const float* __restrict__ g, const float* __restrict__ b,
                   float* __restrict__ out)
{
    __shared__ float s1[256], s2[256];
    int m = blockIdx.x;
    float v[4];
    float sum = 0.f, sq = 0.f;
#pragma unroll
    for (int i = 0; i < 4; ++i) {
        int d = threadIdx.x + i * 256;
        float t = inp[(size_t)m * DM + d];
        if (res) t += res[(size_t)m * DM + d];
        v[i] = t; sum += t; sq += t * t;
    }
    s1[threadIdx.x] = sum; s2[threadIdx.x] = sq;
    __syncthreads();
    for (int off = 128; off > 0; off >>= 1) {
        if (threadIdx.x < off) {
            s1[threadIdx.x] += s1[threadIdx.x + off];
            s2[threadIdx.x] += s2[threadIdx.x + off];
        }
        __syncthreads();
    }
    float mu   = s1[0] / DM;
    float var  = s2[0] / DM - mu * mu;
    float rstd = rsqrtf(var + 1e-5f);
#pragma unroll
    for (int i = 0; i < 4; ++i) {
        int d = threadIdx.x + i * 256;
        out[(size_t)m * DM + d] = (v[i] - mu) * rstd * g[d] + b[d];
    }
}

// ======================================================================
// Host orchestration
// ======================================================================
extern "C" void kernel_launch(void* const* d_in, const int* in_sizes, int n_in,
                              void* d_out, int out_size, void* d_ws, size_t ws_size,
                              hipStream_t stream)
{
    (void)in_sizes; (void)n_in; (void)out_size; (void)ws_size;

    const int*   ids   = (const int*)d_in[0];
    const float* emb   = (const float*)d_in[1];
    const float* inpw  = (const float*)d_in[2];
    const float* convw = (const float*)d_in[3];
    const float* convb = (const float*)d_in[4];
    const float* xprw  = (const float*)d_in[5];
    const float* dtw   = (const float*)d_in[6];
    const float* dtb   = (const float*)d_in[7];
    const float* Alog  = (const float*)d_in[8];
    const float* Dp    = (const float*)d_in[9];
    const float* outw  = (const float*)d_in[10];
    const float* lng   = (const float*)d_in[11];
    const float* lnb   = (const float*)d_in[12];
    const float* fng   = (const float*)d_in[13];
    const float* fnb   = (const float*)d_in[14];
    const float* fcw   = (const float*)d_in[15];
    const float* fcb   = (const float*)d_in[16];
    float* logits = (float*)d_out;

    float* ws = (float*)d_ws;
    size_t o = 0;
    float* x    = ws + o; o += (size_t)MROWS * DM;          // residual stream
    float* xz   = ws + o; o += (size_t)MROWS * 2 * DI;      // in_proj out [xc|z]
    float* xc   = ws + o; o += (size_t)MROWS * DI;          // conv+silu
    float* xdbl = ws + o; o += (size_t)MROWS * (DTRANK + 2 * DSTATE);
    float* dtf  = ws + o; o += (size_t)MROWS * DI;          // softplus(delta)
    float* yb   = ws + o; o += (size_t)MROWS * DI;          // scan output
    float* ob   = ws + o; o += (size_t)MROWS * DM;          // out_proj / final-ln

    auto gemm = [&](const float* A, int lda, const float* W, float* C, int ldc,
                    const float* bias, int M, int N, int K, int mode) {
        dim3 grid((N + BN - 1) / BN, (M + BM - 1) / BM);
        gemm_bf16wmma_kernel<<<grid, 256, 0, stream>>>(A, lda, W, C, ldc,
                                                       bias, M, N, K, mode);
    };

    embed_kernel<<<MROWS, 256, 0, stream>>>(ids, emb, x);

    for (int i = 0; i < NL; ++i) {
        // in_proj: (2048x1024) x (4096x1024)^T -> xz (2048x4096)
        gemm(x, DM, inpw + (size_t)i * 2 * DI * DM, xz, 2 * DI, nullptr,
             MROWS, 2 * DI, DM, 0);
        // depthwise causal conv + bias + silu on the xc half
        conv_silu_kernel<<<dim3(DI / 256, MROWS), 256, 0, stream>>>(
            xz, convw + (size_t)i * DI * DCONV, convb + (size_t)i * DI, xc);
        // x_proj: (2048x2048) x (80x2048)^T -> xdbl (2048x80)
        gemm(xc, DI, xprw + (size_t)i * (DTRANK + 2 * DSTATE) * DI, xdbl,
             DTRANK + 2 * DSTATE, nullptr, MROWS, DTRANK + 2 * DSTATE, DI, 0);
        // dt_proj: (2048x64) x (2048x64)^T + b, softplus -> dtf (2048x2048)
        gemm(xdbl, DTRANK + 2 * DSTATE, dtw + (size_t)i * DI * DTRANK, dtf, DI,
             dtb + (size_t)i * DI, MROWS, DI, DTRANK, 2);
        // selective scan + gating epilogue
        scan_kernel<<<dim3(DI / 256, BB), 256, 0, stream>>>(
            dtf, xdbl, xc, xz, Alog + (size_t)i * DI * DSTATE,
            Dp + (size_t)i * DI, yb);
        // out_proj: (2048x2048) x (1024x2048)^T -> ob (2048x1024)
        gemm(yb, DI, outw + (size_t)i * DM * DI, ob, DM, nullptr,
             MROWS, DM, DI, 0);
        // residual + layernorm -> x
        add_ln_kernel<<<MROWS, 256, 0, stream>>>(
            ob, x, lng + (size_t)i * DM, lnb + (size_t)i * DM, x);
    }

    // final layernorm -> ob
    add_ln_kernel<<<MROWS, 256, 0, stream>>>(x, nullptr, fng, fnb, ob);

    // LM head: (2048x1024) x (32000x1024)^T + fc_b -> logits (2048x32000)
    gemm(ob, DM, fcw, logits, VOCAB_N, fcb, MROWS, VOCAB_N, DM, 1);
}
